// Block_11897059410591
// MI455X (gfx1250) — matrix operand order repaired and checked by reading, not compiled, not run on hardware
//
#include <hip/hip_runtime.h>
#include <hip/hip_bf16.h>

typedef __attribute__((ext_vector_type(16))) _Float16 v16h;
typedef __attribute__((ext_vector_type(8)))  _Float16 v8h;
typedef __attribute__((ext_vector_type(8)))  float    v8f;

#define N_EMBD  1024
#define N_HEAD  16
#define HEAD    64
#define KVR     512
#define TSEQ    2048
#define BATCH   4
#define MROWS   (BATCH * TSEQ)   // 8192
#define FFN     (4 * N_EMBD)     // 4096
#define LN_EPS  1e-5f

// ---------------------------------------------------------------------------
// WMMA helper: D = A(16x32 f16) * B(32x16 f16) + C(16x16 f32), wave32
// ---------------------------------------------------------------------------
__device__ __forceinline__ v8f wmma_f16(v16h a, v16h b, v8f c) {
  return __builtin_amdgcn_wmma_f32_16x16x32_f16(false, a, false, b,
                                                (short)0, c, false, false);
}

// A-fragment (16x32, 16-bit) from row-major f16 [.. x lda]:
// lane L: M = L&15, kb = (L>=16 ? 16 : 0); VGPR0..3 hold K kb..kb+7,
// VGPR4..7 hold K kb+8..kb+15  -> two contiguous 16B loads.
__device__ __forceinline__ v16h load_a_frag(const _Float16* __restrict__ A,
                                            int lda, int m0, int k0, int lane) {
  int m  = m0 + (lane & 15);
  int kb = (lane >> 4) << 4;
  const _Float16* p = A + (size_t)m * lda + k0 + kb;
  v8h lo = *(const v8h*)p;
  v8h hi = *(const v8h*)(p + 8);
  v16h r;
#pragma unroll
  for (int i = 0; i < 8; ++i) { r[i] = lo[i]; r[i + 8] = hi[i]; }
  return r;
}

// Load one K-step's fragments (2 A frags + 4 packed-B frags)
__device__ __forceinline__ void load_frags(
    const _Float16* __restrict__ A, int K, int m0, int k, int lane,
    const _Float16* __restrict__ bbase, int nt0, int ktiles,
    v16h& a0, v16h& a1, v16h bf[4]) {
  a0 = load_a_frag(A, K, m0, k, lane);
  a1 = load_a_frag(A, K, m0 + 16, k, lane);
  int kt = k >> 5;
#pragma unroll
  for (int t = 0; t < 4; ++t)
    bf[t] = *(const v16h*)(bbase + ((size_t)(nt0 + t) * ktiles + kt) * 512);
}

// 8 WMMAs of one K-step (2 M-tiles x 4 N-tiles)
__device__ __forceinline__ void wmma_step(v8f acc[2][4], const v16h& a0,
                                          const v16h& a1, const v16h bf[4]) {
#pragma unroll
  for (int t = 0; t < 4; ++t) {
    acc[0][t] = wmma_f16(a0, bf[t], acc[0][t]);
    acc[1][t] = wmma_f16(a1, bf[t], acc[1][t]);
  }
}

// ---------------------------------------------------------------------------
// Weight pre-pack: f32 row-major [K x N] -> f16 B-fragment tiles.
// Packed layout: tile (nt, kt) at ((nt*(K/32)+kt)*512), lane*16 + e, with
// element (lane, e): K = kt*32 + (lane>=16?16:0) + e, N = nt*16 + (lane&15).
// ---------------------------------------------------------------------------
__global__ __launch_bounds__(256) void pack_b_kernel(
    const float* __restrict__ W, _Float16* __restrict__ Bp, int K, int N) {
  int f = blockIdx.x * 256 + threadIdx.x;           // exact grid, no guard
  int e    = f & 15;
  int lane = (f >> 4) & 31;
  int tile = f >> 9;
  int ktiles = K >> 5;
  int kt = tile % ktiles;
  int nt = tile / ktiles;
  int k = (kt << 5) + ((lane >> 4) << 4) + e;
  int n = (nt << 4) + (lane & 15);
  Bp[f] = (_Float16)W[(size_t)k * N + n];
}

// ---------------------------------------------------------------------------
// Row layernorm: f32 [rows x N] -> f16 [rows x N]
// ---------------------------------------------------------------------------
__global__ __launch_bounds__(128) void ln_rows(
    const float* __restrict__ X, const float* __restrict__ w,
    const float* __restrict__ b, _Float16* __restrict__ Out, int N) {
  __shared__ float s1[128], s2[128];
  int row = blockIdx.x, tid = threadIdx.x;
  const float* x = X + (size_t)row * N;
  float a1 = 0.f, a2 = 0.f;
  for (int i = tid; i < N; i += 128) { float v = x[i]; a1 += v; a2 += v * v; }
  s1[tid] = a1; s2[tid] = a2; __syncthreads();
  for (int off = 64; off > 0; off >>= 1) {
    if (tid < off) { s1[tid] += s1[tid + off]; s2[tid] += s2[tid + off]; }
    __syncthreads();
  }
  float mean = s1[0] / (float)N;
  float var  = s2[0] / (float)N - mean * mean;
  float inv  = rsqrtf(var + LN_EPS);
  _Float16* o = Out + (size_t)row * N;
  for (int i = tid; i < N; i += 128)
    o[i] = (_Float16)((x[i] - mean) * inv * w[i] + b[i]);
}

// ---------------------------------------------------------------------------
// Generic WMMA GEMM: C[MxN] = act(A_f16[MxK] @ Bpacked + bias) + resid
// Block 256 (8 waves, 2x4); wave computes 32(M) x 64(N); block tile 64 x 256.
// Ping-pong software pipeline (2x unrolled K loop, K % 64 == 0): each buffer
// is written by its loads and consumed by its WMMAs directly — no rotation
// copies; compiler emits partial s_wait_loadcnt between WMMA groups.
// ---------------------------------------------------------------------------
__global__ __launch_bounds__(256) void gemm_wmma(
    const _Float16* __restrict__ A, const _Float16* __restrict__ Bp,
    int M, int N, int K,
    const float* __restrict__ bias, const float* __restrict__ resid,
    float* __restrict__ Cf, _Float16* __restrict__ Ch, int act) {
  int lane = threadIdx.x & 31, wid = threadIdx.x >> 5;
  int m0 = blockIdx.y * 64 + (wid & 1) * 32;
  int n0 = blockIdx.x * 256 + (wid >> 1) * 64;
  int ktiles = K >> 5;
  int nt0 = n0 >> 4;
  const _Float16* bbase = Bp + (size_t)lane * 16;

  v8f acc[2][4] = {};
  v16h a0c, a1c, bfc[4];      // "cur" buffer
  v16h a0n, a1n, bfn[4];      // "nxt" buffer

  load_frags(A, K, m0, 0, lane, bbase, nt0, ktiles, a0c, a1c, bfc);

  int nsteps = K >> 6;        // K % 64 == 0 for all call sites
  int k0 = 0;
  for (int it = 0; it < nsteps - 1; ++it, k0 += 64) {
    load_frags(A, K, m0, k0 + 32, lane, bbase, nt0, ktiles, a0n, a1n, bfn);
    wmma_step(acc, a0c, a1c, bfc);
    load_frags(A, K, m0, k0 + 64, lane, bbase, nt0, ktiles, a0c, a1c, bfc);
    wmma_step(acc, a0n, a1n, bfn);
  }
  // tail: k0 == K - 64
  load_frags(A, K, m0, k0 + 32, lane, bbase, nt0, ktiles, a0n, a1n, bfn);
  wmma_step(acc, a0c, a1c, bfc);
  wmma_step(acc, a0n, a1n, bfn);

  int half = lane >> 4, nl = lane & 15;
#pragma unroll
  for (int mt = 0; mt < 2; ++mt) {
#pragma unroll
    for (int t = 0; t < 4; ++t) {
#pragma unroll
      for (int r = 0; r < 8; ++r) {
        int m = m0 + mt * 16 + r + half * 8;
        int n = n0 + t * 16 + nl;
        float v = acc[mt][t][r];
        if (bias)  v += bias[n];
        if (act)   v = 0.5f * v * (1.0f + erff(v * 0.70710678118654752f));
        size_t idx = (size_t)m * N + n;
        if (resid) v += resid[idx];
        if (Cf) Cf[idx] = v;
        if (Ch) Ch[idx] = (_Float16)v;
      }
    }
  }
}

// ---------------------------------------------------------------------------
// RoPE (exact, matches reference: rot = [-x[32:], x[:32]])
// ---------------------------------------------------------------------------
__device__ __forceinline__ float rope_val(float x, float partner, int t, int d) {
  int j = d & 31;
  float ang = (float)t * powf(10000.0f, -(float)j * (1.0f / 32.0f));
  float s, c; sincosf(ang, &s, &c);
  float rot = (d < 32) ? -partner : partner;
  return x * c + rot * s;
}

// Q: f32 [B*T, H*D] -> f16 [B,H,T,D] row-major, with RoPE
__global__ __launch_bounds__(256) void rope_q_pack(
    const float* __restrict__ Q, _Float16* __restrict__ Qh) {
  int f = blockIdx.x * 256 + threadIdx.x;
  int d = f & 63;
  int t = (f >> 6) & (TSEQ - 1);
  int bh = f >> 17;
  int h = bh & (N_HEAD - 1), b = bh >> 4;
  size_t src = (size_t)(b * TSEQ + t) * N_EMBD + h * HEAD;
  float x  = Q[src + d];
  float xp = Q[src + ((d < 32) ? d + 32 : d - 32)];
  Qh[f] = (_Float16)rope_val(x, xp, t, d);
}

// K: from kv f32 [B*T, 2048] cols [0,1024) -> B-fragment tiles (QK^T: k-dim=d)
// per (b,h): tile (st, kt) at (st*2+kt)*512; K = d, N = key index.
__global__ __launch_bounds__(256) void pack_k_rope(
    const float* __restrict__ KV, _Float16* __restrict__ Kp) {
  int f = blockIdx.x * 256 + threadIdx.x;
  int e    = f & 15;
  int lane = (f >> 4) & 31;
  int r    = f >> 9;
  int kt = r & 1;          r >>= 1;
  int st = r & 127;        r >>= 7;          // T/16 = 128
  int h  = r & (N_HEAD - 1);
  int b  = r >> 4;
  int s  = st * 16 + (lane & 15);
  int dd = kt * 32 + ((lane >> 4) << 4) + e;
  size_t base = (size_t)(b * TSEQ + s) * (2 * N_EMBD) + h * HEAD;
  float x  = KV[base + dd];
  float xp = KV[base + ((dd < 32) ? dd + 32 : dd - 32)];
  Kp[f] = (_Float16)rope_val(x, xp, s, dd);
}

// V: from kv f32 cols [1024,2048) -> B-fragment tiles (PV: k-dim = key)
// per (b,h): tile (dt, st) at (dt*(T/32)+st)*512; K = key, N = d.
__global__ __launch_bounds__(256) void pack_v(
    const float* __restrict__ KV, _Float16* __restrict__ Vp) {
  int f = blockIdx.x * 256 + threadIdx.x;
  int e    = f & 15;
  int lane = (f >> 4) & 31;
  int r    = f >> 9;
  int st = r & 63;         r >>= 6;          // T/32 = 64
  int dt = r & 3;          r >>= 2;
  int h  = r & (N_HEAD - 1);
  int b  = r >> 4;
  int s  = st * 32 + ((lane >> 4) << 4) + e;
  int dd = dt * 16 + (lane & 15);
  Vp[f] = (_Float16)KV[(size_t)(b * TSEQ + s) * (2 * N_EMBD) + N_EMBD
                       + h * HEAD + dd];
}

// ---------------------------------------------------------------------------
// Flash attention, causal. Wave = 16 query rows; block 256 = 128 queries.
// grid = (T/128, H, B). Online softmax in WMMA C-layout; P goes through a
// per-wave 1KB LDS slab to re-enter A-layout for the P@V WMMAs.
// ---------------------------------------------------------------------------
__global__ __launch_bounds__(256) void flash_attn(
    const _Float16* __restrict__ Qh, const _Float16* __restrict__ Kp,
    const _Float16* __restrict__ Vp, _Float16* __restrict__ Oh) {
  __shared__ _Float16 plds[8 * 512];
  int lane = threadIdx.x & 31, wid = threadIdx.x >> 5;
  int q0 = blockIdx.x * 128 + wid * 16;
  int h = blockIdx.y, b = blockIdx.z;
  size_t bh = (size_t)(b * N_HEAD + h);
  const _Float16* Qb = Qh + bh * TSEQ * HEAD;
  const _Float16* Kb = Kp + bh * (TSEQ / 16) * 2 * 512 + (size_t)lane * 16;
  const _Float16* Vb = Vp + bh * 4 * (TSEQ / 32) * 512 + (size_t)lane * 16;

  v16h qa0 = load_a_frag(Qb, HEAD, q0, 0, lane);
  v16h qa1 = load_a_frag(Qb, HEAD, q0, 32, lane);

  v8f o[4] = {};
  float mi[8], li[8];
#pragma unroll
  for (int r = 0; r < 8; ++r) { mi[r] = -3.0e38f; li[r] = 0.f; }
  int half = lane >> 4, nl = lane & 15;
  _Float16* pw = plds + wid * 512;
  const _Float16* pr = plds + wid * 512 + (lane & 15) * 32 + (half << 4);

  int nchunks = (q0 + 15) / 32 + 1;
  for (int c = 0; c < nchunks; ++c) {
    int s0 = c * 32;
    v16h k00 = *(const v16h*)(Kb + (size_t)(4 * c + 0) * 512);
    v16h k01 = *(const v16h*)(Kb + (size_t)(4 * c + 1) * 512);
    v16h k10 = *(const v16h*)(Kb + (size_t)(4 * c + 2) * 512);
    v16h k11 = *(const v16h*)(Kb + (size_t)(4 * c + 3) * 512);
    v8f sA = {}, sB = {};
    sA = wmma_f16(qa0, k00, sA);
    sA = wmma_f16(qa1, k01, sA);
    sB = wmma_f16(qa0, k10, sB);
    sB = wmma_f16(qa1, k11, sB);
#pragma unroll
    for (int r = 0; r < 8; ++r) {
      int row = r + half * 8;
      int qg = q0 + row;
      float a0 = sA[r] * 0.125f;              // 1/sqrt(64)
      float a1 = sB[r] * 0.125f;
      if (s0 + nl > qg)      a0 = -3.0e38f;   // causal mask
      if (s0 + 16 + nl > qg) a1 = -3.0e38f;
      float rm = fmaxf(a0, a1);
      rm = fmaxf(rm, __shfl_xor(rm, 1));
      rm = fmaxf(rm, __shfl_xor(rm, 2));
      rm = fmaxf(rm, __shfl_xor(rm, 4));
      rm = fmaxf(rm, __shfl_xor(rm, 8));
      float mnew  = fmaxf(mi[r], rm);
      float alpha = __expf(mi[r] - mnew);
      float p0 = __expf(a0 - mnew);
      float p1 = __expf(a1 - mnew);
      float rs = p0 + p1;
      rs += __shfl_xor(rs, 1);
      rs += __shfl_xor(rs, 2);
      rs += __shfl_xor(rs, 4);
      rs += __shfl_xor(rs, 8);
      li[r] = li[r] * alpha + rs;
      mi[r] = mnew;
#pragma unroll
      for (int t = 0; t < 4; ++t) o[t][r] *= alpha;
      pw[row * 32 + nl]      = (_Float16)p0;  // C-layout -> LDS (row-major)
      pw[row * 32 + 16 + nl] = (_Float16)p1;
    }
    v8h lo = *(const v8h*)pr;                 // reload as A-fragment
    v8h hi = *(const v8h*)(pr + 8);
    v16h pa;
#pragma unroll
    for (int i = 0; i < 8; ++i) { pa[i] = lo[i]; pa[i + 8] = hi[i]; }
#pragma unroll
    for (int t = 0; t < 4; ++t) {
      v16h vb = *(const v16h*)(Vb + (size_t)(t * (TSEQ / 32) + c) * 512);
      o[t] = wmma_f16(pa, vb, o[t]);
    }
  }
#pragma unroll
  for (int r = 0; r < 8; ++r) {
    int row = r + half * 8;
    float inv = 1.0f / li[r];
    size_t base = (size_t)(b * TSEQ + q0 + row) * N_EMBD + h * HEAD;
#pragma unroll
    for (int t = 0; t < 4; ++t)
      Oh[base + t * 16 + nl] = (_Float16)(o[t][r] * inv);
  }
}

// ---------------------------------------------------------------------------
extern "C" void kernel_launch(void* const* d_in, const int* in_sizes, int n_in,
                              void* d_out, int out_size, void* d_ws,
                              size_t ws_size, hipStream_t stream) {
  (void)in_sizes; (void)n_in; (void)out_size; (void)ws_size;
  const float* x      = (const float*)d_in[0];
  const float* ln1_w  = (const float*)d_in[1];
  const float* ln1_b  = (const float*)d_in[2];
  const float* ln2_w  = (const float*)d_in[3];
  const float* ln2_b  = (const float*)d_in[4];
  const float* q_w    = (const float*)d_in[5];
  const float* kvd_w  = (const float*)d_in[6];
  const float* kvln_w = (const float*)d_in[7];
  const float* kvln_b = (const float*)d_in[8];
  const float* kvu_w  = (const float*)d_in[9];
  const float* proj_w = (const float*)d_in[10];
  const float* proj_b = (const float*)d_in[11];
  const float* f1_w   = (const float*)d_in[12];
  const float* f1_b   = (const float*)d_in[13];
  const float* f2_w   = (const float*)d_in[14];
  const float* f2_b   = (const float*)d_in[15];

  char* ws = (char*)d_ws;
  const size_t MB = 1024 * 1024;
  // workspace map (total 207 MiB, with aliasing)
  const size_t OFF_WQ    = 0;                 // 2 MiB  f16 packed [1024x1024]
  const size_t OFF_WKVD  = OFF_WQ    + 2*MB;  // 1 MiB  f16 packed [1024x512]
  const size_t OFF_WKVU  = OFF_WKVD  + 1*MB;  // 2 MiB  f16 packed [512x2048]
  const size_t OFF_WPROJ = OFF_WKVU  + 2*MB;  // 2 MiB  f16 packed [1024x1024]
  const size_t OFF_WF1   = OFF_WPROJ + 2*MB;  // 8 MiB  f16 packed [1024x4096]
  const size_t OFF_WF2   = OFF_WF1   + 8*MB;  // 8 MiB  f16 packed [4096x1024]
  const size_t OFF_HA    = OFF_WF2   + 8*MB;  // 16 MiB f16 [8192,1024]; reused as attn-out
  const size_t OFF_QF32  = OFF_HA    + 16*MB; // 32 MiB f32 [8192,1024]; reused as X2
  const size_t OFF_CKVR  = OFF_QF32  + 32*MB; // 16 MiB f32 [8192,512];  reused as H2 f16
  const size_t OFF_CKVH  = OFF_CKVR  + 16*MB; //  8 MiB f16 [8192,512]
  const size_t OFF_KV    = OFF_CKVH  + 8*MB;  // 64 MiB f32 [8192,2048]; reused as FF1 f16
  const size_t OFF_QH    = OFF_KV    + 64*MB; // 16 MiB f16 [B,H,T,64]
  const size_t OFF_KP    = OFF_QH    + 16*MB; // 16 MiB f16 packed K frags
  const size_t OFF_VP    = OFF_KP    + 16*MB; // 16 MiB f16 packed V frags

  _Float16* WQ    = (_Float16*)(ws + OFF_WQ);
  _Float16* WKVD  = (_Float16*)(ws + OFF_WKVD);
  _Float16* WKVU  = (_Float16*)(ws + OFF_WKVU);
  _Float16* WPROJ = (_Float16*)(ws + OFF_WPROJ);
  _Float16* WF1   = (_Float16*)(ws + OFF_WF1);
  _Float16* WF2   = (_Float16*)(ws + OFF_WF2);
  _Float16* HA    = (_Float16*)(ws + OFF_HA);
  float*    QF32  = (float*)   (ws + OFF_QF32);
  float*    CKVR  = (float*)   (ws + OFF_CKVR);
  _Float16* CKVH  = (_Float16*)(ws + OFF_CKVH);
  float*    KV    = (float*)   (ws + OFF_KV);
  _Float16* QH    = (_Float16*)(ws + OFF_QH);
  _Float16* KP    = (_Float16*)(ws + OFF_KP);
  _Float16* VP    = (_Float16*)(ws + OFF_VP);
  _Float16* ATTNH = HA;                         // alias (HA dead after kvd GEMM)
  float*    X2    = QF32;                       // alias (QF32 dead after rope)
  _Float16* H2    = (_Float16*)CKVR;            // alias (CKVR dead after kv-LN)
  _Float16* FF1H  = (_Float16*)KV;              // alias (KV dead after pack)

  // 1) pack weights f32 -> f16 WMMA B-fragment layout
  pack_b_kernel<<<(N_EMBD*N_EMBD)/256, 256, 0, stream>>>(q_w,    WQ,    N_EMBD, N_EMBD);
  pack_b_kernel<<<(N_EMBD*KVR)/256,    256, 0, stream>>>(kvd_w,  WKVD,  N_EMBD, KVR);
  pack_b_kernel<<<(KVR*2*N_EMBD)/256,  256, 0, stream>>>(kvu_w,  WKVU,  KVR,    2*N_EMBD);
  pack_b_kernel<<<(N_EMBD*N_EMBD)/256, 256, 0, stream>>>(proj_w, WPROJ, N_EMBD, N_EMBD);
  pack_b_kernel<<<(N_EMBD*FFN)/256,    256, 0, stream>>>(f1_w,   WF1,   N_EMBD, FFN);
  pack_b_kernel<<<(FFN*N_EMBD)/256,    256, 0, stream>>>(f2_w,   WF2,   FFN,    N_EMBD);

  // 2) h = LN1(x) -> f16
  ln_rows<<<MROWS, 128, 0, stream>>>(x, ln1_w, ln1_b, HA, N_EMBD);

  // 3) q = h @ q_w ; 4) ckv_raw = h @ kvd_w
  gemm_wmma<<<dim3(N_EMBD/256, MROWS/64), 256, 0, stream>>>(
      HA, WQ, MROWS, N_EMBD, N_EMBD, nullptr, nullptr, QF32, nullptr, 0);
  gemm_wmma<<<dim3(KVR/256, MROWS/64), 256, 0, stream>>>(
      HA, WKVD, MROWS, KVR, N_EMBD, nullptr, nullptr, CKVR, nullptr, 0);

  // 5) ckv = LN(ckv_raw) -> f16 ; 6) kv = ckv @ kvu_w
  ln_rows<<<MROWS, 128, 0, stream>>>(CKVR, kvln_w, kvln_b, CKVH, KVR);
  gemm_wmma<<<dim3((2*N_EMBD)/256, MROWS/64), 256, 0, stream>>>(
      CKVH, WKVU, MROWS, 2*N_EMBD, KVR, nullptr, nullptr, KV, nullptr, 0);

  // 7) RoPE + fragment packing for attention
  rope_q_pack<<<(BATCH*N_HEAD*TSEQ*HEAD)/256, 256, 0, stream>>>(QF32, QH);
  pack_k_rope<<<(BATCH*N_HEAD*(TSEQ/16)*2*512)/256, 256, 0, stream>>>(KV, KP);
  pack_v     <<<(BATCH*N_HEAD*4*(TSEQ/32)*512)/256, 256, 0, stream>>>(KV, VP);

  // 8) causal flash attention
  flash_attn<<<dim3(TSEQ/128, N_HEAD, BATCH), 256, 0, stream>>>(QH, KP, VP, ATTNH);

  // 9) x2 = x + attn @ proj_w + proj_b
  gemm_wmma<<<dim3(N_EMBD/256, MROWS/64), 256, 0, stream>>>(
      ATTNH, WPROJ, MROWS, N_EMBD, N_EMBD, proj_b, x, X2, nullptr, 0);

  // 10) h2 = LN2(x2) ; 11) ff1 = gelu(h2 @ f1_w + f1_b) -> f16
  ln_rows<<<MROWS, 128, 0, stream>>>(X2, ln2_w, ln2_b, H2, N_EMBD);
  gemm_wmma<<<dim3(FFN/256, MROWS/64), 256, 0, stream>>>(
      H2, WF1, MROWS, FFN, N_EMBD, f1_b, nullptr, nullptr, FF1H, 1);

  // 12) out = x2 + ff1 @ f2_w + f2_b
  gemm_wmma<<<dim3(N_EMBD/256, MROWS/64), 256, 0, stream>>>(
      FF1H, WF2, MROWS, N_EMBD, FFN, f2_b, X2, (float*)d_out, nullptr, 0);
}
